// MSEF_9363028706105
// MI455X (gfx1250) — matrix operand order repaired and checked by reading, not compiled, not run on hardware
//
#include <hip/hip_runtime.h>

typedef __attribute__((ext_vector_type(16))) _Float16 v16h;
typedef __attribute__((ext_vector_type(8)))  float    v8f;

#define C_CH 256
#define HW   4096
#define TP   32      // pixels per tile
#define LDA  258     // padded f16 row stride (bank-conflict free frag reads)
#define LDO  34      // padded f32 transpose-out stride

__device__ __forceinline__ float sigf(float x) { return 1.f / (1.f + expf(-x)); }

// ---------------------------------------------------------------------------
// WMMA GEMM over one 32-pixel tile.
// sA: LDS f16 [32 px][LDA], col = input channel k (0..255)
// Wg: global f32 weights [256 out][256 in]
// 8 waves: wave&1 -> M-tile (16 px), wave>>1 -> 64 output channels (4 N-tiles)
// acc[4]: f32 16x16 accumulators (v8f each), one per N-tile.
// ---------------------------------------------------------------------------
__device__ __forceinline__ void gemm_tile(const _Float16* sA, _Float16* sB,
                                          const float* __restrict__ Wg, v8f acc[4]) {
  const int t = threadIdx.x;
  const int lane = t & 31, wave = t >> 5;
  const int wm = wave & 1, wn = wave >> 1;
  const int m = lane & 15, hf = lane >> 4;
  for (int kc = 0; kc < 8; ++kc) {
    __syncthreads();
    // stage B chunk: B[k][n] = Wg[n][kc*32+k], k = lane (coalesced 128B reads)
    for (int nn = 0; nn < 32; ++nn) {
      int nch = nn * 8 + wave;
      sB[lane * LDA + nch] = (_Float16)Wg[(size_t)nch * 256 + kc * 32 + lane];
    }
    __syncthreads();
    // A fragment: 16-bit A 16x32 layout (ISA 7.12.2)
    union { v16h v; unsigned u[8]; } af;
#pragma unroll
    for (int p = 0; p < 8; ++p) {
      int k0 = (p < 4) ? (2 * p + 8 * hf) : (16 + 2 * (p - 4) + 8 * hf);
      af.u[p] = *(const unsigned*)(sA + (16 * wm + m) * LDA + kc * 32 + k0);
    }
    const int kb = m + 16 * hf;  // B row (K within chunk)
#pragma unroll
    for (int uu = 0; uu < 4; ++uu) {
      union { v16h v; unsigned u[8]; } bf;
#pragma unroll
      for (int s = 0; s < 16; ++s)
        bf.v[s] = sB[((s < 8) ? (s + 8 * hf) : (16 + (s - 8) + 8 * hf)) * LDA + 64 * wn + 16 * uu + m];
      acc[uu] = __builtin_amdgcn_wmma_f32_16x16x32_f16(
          false, af.v, false, bf.v, (short)0, acc[uu], false, false);
    }
  }
  __syncthreads();  // protect LDS reuse after the GEMM
}

// Transpose accumulators through LDS for coalesced channel-major global stores.
// dst points at element (n, c=0, p0). vals[u][r] laid out per C-matrix layout.
__device__ __forceinline__ void transpose_store(float* sOut, float* __restrict__ dst,
                                                const float (*vals)[8]) {
  const int t = threadIdx.x;
  const int lane = t & 31, wave = t >> 5;
  const int wm = wave & 1, wn = wave >> 1;
  const int m = lane & 15, hf = lane >> 4;
  for (int half = 0; half < 2; ++half) {
    __syncthreads();
    if ((wn >> 1) == half) {
#pragma unroll
      for (int uu = 0; uu < 4; ++uu) {
        int c_loc = 64 * wn - 128 * half + 16 * uu + m;
#pragma unroll
        for (int r = 0; r < 8; ++r) {
          int px = 16 * wm + r + 8 * hf;
          sOut[c_loc * LDO + px] = vals[uu][r];
        }
      }
    }
    __syncthreads();
    for (int cc = 0; cc < 16; ++cc) {
      int c_loc = cc * 8 + wave;
      dst[(size_t)(128 * half + c_loc) * HW + lane] = sOut[c_loc * LDO + lane];
    }
  }
}

// ---------------------------------------------------------------------------
// K1: g1 = conv1(x1)+b1 ; XG = g1 + conv2(x2)+b2 ; GAP + group-stat reductions
// ---------------------------------------------------------------------------
__global__ __launch_bounds__(256) void k1_gemm12(
    const float* __restrict__ x1, const float* __restrict__ x2,
    const float* __restrict__ w1c, const float* __restrict__ b1c,
    const float* __restrict__ w2c, const float* __restrict__ b2c,
    float* __restrict__ g1_buf, float* __restrict__ xg_buf,
    float* __restrict__ gap_sum, float* __restrict__ gsum, float* __restrict__ gsq) {
  __shared__ __align__(16) unsigned char smem[2 * TP * LDA * 2];  // 33024 B
  _Float16* sA = (_Float16*)smem;
  _Float16* sB = (_Float16*)(smem + TP * LDA * 2);
  float* sOut = (float*)smem;  // reused after GEMMs

  const int b = blockIdx.x;
  const int n = b >> 7, p0 = (b & 127) * TP;
  const size_t base = (size_t)n * C_CH * HW + p0;
  const int t = threadIdx.x, lane = t & 31, wave = t >> 5;

  for (int cc = 0; cc < 32; ++cc) {
    int c = cc * 8 + wave;
    sA[lane * LDA + c] = (_Float16)x1[base + (size_t)c * HW + lane];
  }
  v8f z = {0.f, 0.f, 0.f, 0.f, 0.f, 0.f, 0.f, 0.f};
  v8f acc1[4] = {z, z, z, z};
  gemm_tile(sA, sB, w1c, acc1);

  for (int cc = 0; cc < 32; ++cc) {
    int c = cc * 8 + wave;
    sA[lane * LDA + c] = (_Float16)x2[base + (size_t)c * HW + lane];
  }
  v8f acc2[4] = {z, z, z, z};
  gemm_tile(sA, sB, w2c, acc2);

  const int wn = wave >> 1, m = lane & 15;
  float g1v[4][8], xgv[4][8];
#pragma unroll
  for (int uu = 0; uu < 4; ++uu) {
    int c = 64 * wn + 16 * uu + m;
    float bb1 = b1c[c], bb2 = b2c[c];
    float s = 0.f, q = 0.f;
#pragma unroll
    for (int r = 0; r < 8; ++r) {
      float g1e = acc1[uu][r] + bb1;
      float xge = g1e + acc2[uu][r] + bb2;
      g1v[uu][r] = g1e;
      xgv[uu][r] = xge;
      s += xge;
      q += xge * xge;
    }
    s += __shfl_xor(s, 16);  // combine pixel halves (same channel)
    q += __shfl_xor(q, 16);
    if (lane < 16) atomicAdd(&gap_sum[n * C_CH + c], s);
#pragma unroll
    for (int off = 8; off >= 1; off >>= 1) {
      s += __shfl_xor(s, off);
      q += __shfl_xor(q, off);
    }
    if (lane == 0) {
      atomicAdd(&gsum[n * 16 + 4 * wn + uu], s);
      atomicAdd(&gsq[n * 16 + 4 * wn + uu], q);
    }
  }
  transpose_store(sOut, g1_buf + base, g1v);
  transpose_store(sOut, xg_buf + base, xgv);
}

// ---------------------------------------------------------------------------
// K2: rw, channel-att sigmoid, group mu/inv-sd, BN scale/shift precompute
// ---------------------------------------------------------------------------
__global__ __launch_bounds__(256) void k2_small(
    const float* __restrict__ gap_sum, const float* __restrict__ gsum,
    const float* __restrict__ gsq, const float* __restrict__ ca_w,
    const float* __restrict__ bn_g, const float* __restrict__ bn_b,
    const float* __restrict__ bn_rm, const float* __restrict__ bn_rv,
    float* rw, float* sca, float* bns, float* bnb2, float* mu_g, float* isd_g) {
  int n = blockIdx.x, c = threadIdx.x;
  const float inv = 1.f / 4096.f;
  float gm = gap_sum[n * C_CH + c] * inv;
  rw[n * C_CH + c] = sigf(gm);
  float a = 0.f;
#pragma unroll
  for (int j = 0; j < 5; ++j) {
    int cc = c - 2 + j;
    if (cc >= 0 && cc < C_CH) a += ca_w[j] * gap_sum[n * C_CH + cc] * inv;
  }
  sca[n * C_CH + c] = sigf(a);
  if (c < 16) {
    float s = gsum[n * 16 + c], q = gsq[n * 16 + c];
    const float cnt = 65536.f;
    float mu = s / cnt;
    float var = (q - s * s / cnt) / (cnt - 1.f);
    var = var > 0.f ? var : 0.f;
    mu_g[n * 16 + c] = mu;
    isd_g[n * 16 + c] = 1.f / (sqrtf(var) + 1e-10f);
  }
  if (n == 0) {
    float iv = bn_g[c] * rsqrtf(bn_rv[c] + 1e-5f);
    bns[c] = iv;
    bnb2[c] = bn_b[c] - bn_rm[c] * iv;
  }
}

// ---------------------------------------------------------------------------
// K3: threshold masks -> x_low (in place over g1) + per-(n,c) x_low sums
// ---------------------------------------------------------------------------
__global__ __launch_bounds__(256) void k3_masks(
    const float* __restrict__ xg_buf, float* __restrict__ g1low,
    const float* __restrict__ rw, const float* __restrict__ bns,
    const float* __restrict__ bnb2, float* __restrict__ low_sum) {
  __shared__ float red[256];
  int b = blockIdx.x, c = b & 255, t = threadIdx.x;
  size_t base = (size_t)b * HW;
  float sc = bns[c], sh = bnb2[c], rwv = rw[b];
  float s = 0.f;
  for (int i = 0; i < 16; ++i) {
    int px = t + i * 256;
    float xg = xg_buf[base + px];
    float g1 = g1low[base + px];
    float w1 = sigf(g1 * sc + sh);
    float w2 = sigf((xg - g1) * sc + sh);
    float f = (rwv < w1 ? 1.f : 0.f) + (rwv < w2 ? 1.f : 0.f);
    float xl = xg * f;
    g1low[base + px] = xl;
    s += xl;
  }
  red[t] = s;
  __syncthreads();
  for (int o = 128; o > 0; o >>= 1) {
    if (t < o) red[t] += red[t + o];
    __syncthreads();
  }
  if (t == 0) low_sum[b] = red[0];
}

// ---------------------------------------------------------------------------
// K4: gate = softmax(relu(low_mean @ gate_w^T + b))
// ---------------------------------------------------------------------------
__global__ __launch_bounds__(256) void k4_gate(
    const float* __restrict__ low_sum, const float* __restrict__ gate_w,
    const float* __restrict__ gate_b, float* __restrict__ gate) {
  __shared__ float red[256];
  int n = blockIdx.x, c = threadIdx.x;
  float a = gate_b[c];
  for (int k = 0; k < C_CH; ++k)
    a += (low_sum[n * C_CH + k] * (1.f / 4096.f)) * gate_w[c * C_CH + k];
  a = a > 0.f ? a : 0.f;
  red[c] = a;
  __syncthreads();
  for (int o = 128; o > 0; o >>= 1) {
    if (c < o) red[c] = fmaxf(red[c], red[c + o]);
    __syncthreads();
  }
  float vmax = red[0];
  __syncthreads();
  float e = expf(a - vmax);
  red[c] = e;
  __syncthreads();
  for (int o = 128; o > 0; o >>= 1) {
    if (c < o) red[c] += red[c + o];
    __syncthreads();
  }
  gate[n * C_CH + c] = e / red[0];
}

// ---------------------------------------------------------------------------
// K5: fused main pass -> xL (written to d_out as scratch)
// ---------------------------------------------------------------------------
__global__ __launch_bounds__(256) void k5_main(
    const float* __restrict__ xg_buf, const float* __restrict__ xlow_buf,
    const float* __restrict__ w3c, const float* __restrict__ b3c,
    const float* __restrict__ w4c, const float* __restrict__ b4c,
    const float* __restrict__ c4g_w, const float* __restrict__ c4g_b,
    const float* __restrict__ dww, const float* __restrict__ dwb,
    const float* __restrict__ gate, const float* __restrict__ sca,
    const float* __restrict__ mu_g, const float* __restrict__ isd_g,
    const float* __restrict__ gamma, const float* __restrict__ beta,
    float* __restrict__ out) {
  __shared__ __align__(16) unsigned char smem[3 * TP * LDA * 2 + 128];  // 49664 B
  _Float16* sXG = (_Float16*)smem;
  _Float16* sA = (_Float16*)(smem + TP * LDA * 2);
  _Float16* sB = (_Float16*)(smem + 2 * TP * LDA * 2);
  float* sOut = (float*)(smem + TP * LDA * 2);  // overlays sA/sB after GEMMs
  float* spix = (float*)(smem + 3 * TP * LDA * 2);

  int b = blockIdx.x;
  int n = b >> 7, p0 = (b & 127) * TP;
  size_t base = (size_t)n * C_CH * HW + p0;
  int t = threadIdx.x, lane = t & 31, wave = t >> 5;
  int h = p0 >> 6, w0 = p0 & 63;

  for (int cc = 0; cc < 32; ++cc) {
    int c = cc * 8 + wave;
    sXG[lane * LDA + c] = (_Float16)xg_buf[base + (size_t)c * HW + lane];
  }
  __syncthreads();
  if (t < 32) {  // per-pixel spatial gate sigma(conv4g . XG + b)
    float a = c4g_b[0];
    for (int c = 0; c < C_CH; ++c) a += c4g_w[c] * (float)sXG[t * LDA + c];
    spix[t] = sigf(a);
  }
  // A = x_up = 2*XG - x_low
  for (int cc = 0; cc < 32; ++cc) {
    int c = cc * 8 + wave;
    float xg = (float)sXG[lane * LDA + c];
    float xl = xlow_buf[base + (size_t)c * HW + lane];
    sA[lane * LDA + c] = (_Float16)(2.f * xg - xl);
  }
  v8f z = {0.f, 0.f, 0.f, 0.f, 0.f, 0.f, 0.f, 0.f};
  v8f accU[4] = {z, z, z, z};
  gemm_tile(sA, sB, w4c, accU);

  // A = depthwise3x3(x_low) + dw_b  (zero-padded)
  for (int cc = 0; cc < 32; ++cc) {
    int c = cc * 8 + wave;
    const float* wt = dww + c * 9;
    const float* rowp = xlow_buf + (size_t)(n * C_CH + c) * HW;
    float a = dwb[c];
    int wimg = w0 + lane;
#pragma unroll
    for (int dy = -1; dy <= 1; ++dy) {
      int hh = h + dy;
      if (hh < 0 || hh >= 64) continue;
#pragma unroll
      for (int dx = -1; dx <= 1; ++dx) {
        int ww = wimg + dx;
        if (ww < 0 || ww >= 64) continue;
        a += wt[(dy + 1) * 3 + (dx + 1)] * rowp[hh * 64 + ww];
      }
    }
    sA[lane * LDA + c] = (_Float16)a;
  }
  v8f accX[4] = {z, z, z, z};
  gemm_tile(sA, sB, w3c, accX);

  // fused epilogue: xL = (x11+b3)*gate + (up+b4) + x_gui + temp
  int wm = wave & 1, wn = wave >> 1, m = lane & 15, hf = lane >> 4;
  float vals[4][8];
#pragma unroll
  for (int uu = 0; uu < 4; ++uu) {
    int c = 64 * wn + 16 * uu + m;
    float gv = gate[n * C_CH + c], sv = sca[n * C_CH + c];
    float gm = mu_g[n * 16 + (c >> 4)], gi = isd_g[n * 16 + (c >> 4)];
    float gam = gamma[c], bet = beta[c];
    float bb3 = b3c[c], bb4 = b4c[c];
#pragma unroll
    for (int r = 0; r < 8; ++r) {
      int px = 16 * wm + r + 8 * hf;
      float xg = (float)sXG[px * LDA + c];
      float xs = spix[px] * xg;
      float xgui = (xs - gm) * gi * gam + bet;
      float temp = xg * sv;
      vals[uu][r] = (accX[uu][r] + bb3) * gv + (accU[uu][r] + bb4) + xgui + temp;
    }
  }
  transpose_store(sOut, out + base, vals);
}

// ---------------------------------------------------------------------------
// K6: H-pool and W-pool of xL
// ---------------------------------------------------------------------------
__global__ __launch_bounds__(256) void k6_pool(const float* __restrict__ xl,
                                               float* __restrict__ xh,
                                               float* __restrict__ xw) {
  __shared__ float hs[64], wsm[64];
  int b = blockIdx.x, t = threadIdx.x;
  size_t base = (size_t)b * HW;
  if (t < 64) { hs[t] = 0.f; wsm[t] = 0.f; }
  __syncthreads();
  for (int i = 0; i < 16; ++i) {
    int px = t + i * 256;
    float v = xl[base + px];
    atomicAdd(&hs[px >> 6], v);
    atomicAdd(&wsm[px & 63], v);
  }
  __syncthreads();
  if (t < 64) xh[(size_t)b * 64 + t] = hs[t] * (1.f / 64.f);
  else if (t < 128) xw[(size_t)b * 64 + (t - 64)] = wsm[t - 64] * (1.f / 64.f);
}

// ---------------------------------------------------------------------------
// K7: coordinate attention -> sh[n,c,h], sw[n,c,w]
// ---------------------------------------------------------------------------
__global__ __launch_bounds__(256) void k7_coord(
    const float* __restrict__ xh, const float* __restrict__ xw,
    const float* __restrict__ la1_w, const float* __restrict__ lbg,
    const float* __restrict__ lbb, const float* __restrict__ lbrm,
    const float* __restrict__ lbrv, const float* __restrict__ fh_w,
    const float* __restrict__ fw_w, float* __restrict__ sh, float* __restrict__ sw) {
  __shared__ float yc[16 * 128];
  int n = blockIdx.x, t = threadIdx.x;
  for (int j = 0; j < 8; ++j) {
    int idx = t + j * 256;
    int mm = idx >> 7, pos = idx & 127;
    const float* wrow = la1_w + mm * C_CH;
    float a = 0.f;
    for (int c = 0; c < C_CH; ++c) {
      float pv = (pos < 64) ? xh[((size_t)n * C_CH + c) * 64 + pos]
                            : xw[((size_t)n * C_CH + c) * 64 + pos - 64];
      a += wrow[c] * pv;
    }
    float iv = lbg[mm] * rsqrtf(lbrv[mm] + 1e-5f);
    float v = a * iv + (lbb[mm] - lbrm[mm] * iv);
    yc[mm * 128 + pos] = v > 0.f ? v : 0.f;
  }
  __syncthreads();
  for (int j = 0; j < 64; ++j) {
    int idx = t + j * 256;
    int c = idx >> 6, hh = idx & 63;
    float a = 0.f, a2 = 0.f;
#pragma unroll
    for (int mm = 0; mm < 16; ++mm) {
      a += fh_w[c * 16 + mm] * yc[mm * 128 + hh];
      a2 += fw_w[c * 16 + mm] * yc[mm * 128 + 64 + hh];
    }
    sh[((size_t)n * C_CH + c) * 64 + hh] = sigf(a);
    sw[((size_t)n * C_CH + c) * 64 + hh] = sigf(a2);
  }
}

// ---------------------------------------------------------------------------
// K8: out = xL * sh * sw  (in place on d_out)
// ---------------------------------------------------------------------------
__global__ __launch_bounds__(256) void k8_final(float* __restrict__ out,
                                                const float* __restrict__ sh,
                                                const float* __restrict__ sw) {
  int b = blockIdx.x, t = threadIdx.x;
  size_t base = (size_t)b * HW;
  for (int i = 0; i < 16; ++i) {
    int px = t + i * 256;
    out[base + px] *= sh[(size_t)b * 64 + (px >> 6)] * sw[(size_t)b * 64 + (px & 63)];
  }
}

__global__ void kz_zero(float* p, int nfl) {
  int i = blockIdx.x * 256 + threadIdx.x;
  if (i < nfl) p[i] = 0.f;
}

// ---------------------------------------------------------------------------
extern "C" void kernel_launch(void* const* d_in, const int* in_sizes, int n_in,
                              void* d_out, int out_size, void* d_ws, size_t ws_size,
                              hipStream_t stream) {
  (void)in_sizes; (void)n_in; (void)out_size; (void)ws_size;
  const float* x1     = (const float*)d_in[0];
  const float* x2     = (const float*)d_in[1];
  const float* c1w    = (const float*)d_in[2];
  const float* c1b    = (const float*)d_in[3];
  const float* c2w    = (const float*)d_in[4];
  const float* c2b    = (const float*)d_in[5];
  const float* c3w    = (const float*)d_in[6];
  const float* c3b    = (const float*)d_in[7];
  const float* c4w    = (const float*)d_in[8];
  const float* c4b    = (const float*)d_in[9];
  const float* c4gw   = (const float*)d_in[10];
  const float* c4gb   = (const float*)d_in[11];
  const float* bng    = (const float*)d_in[12];
  const float* bnb    = (const float*)d_in[13];
  const float* bnrm   = (const float*)d_in[14];
  const float* bnrv   = (const float*)d_in[15];
  const float* gamma  = (const float*)d_in[16];
  const float* beta   = (const float*)d_in[17];
  const float* gatew  = (const float*)d_in[18];
  const float* gateb  = (const float*)d_in[19];
  const float* dww    = (const float*)d_in[20];
  const float* dwb    = (const float*)d_in[21];
  // d_in[22]=interact_w, d_in[23]=interact_b: dead code (softmax over size-1 axis == 1)
  const float* caw    = (const float*)d_in[24];
  const float* la1w   = (const float*)d_in[25];
  const float* lbg    = (const float*)d_in[26];
  const float* lbb    = (const float*)d_in[27];
  const float* lbrm   = (const float*)d_in[28];
  const float* lbrv   = (const float*)d_in[29];
  const float* fhw    = (const float*)d_in[30];
  const float* fww    = (const float*)d_in[31];
  float* out = (float*)d_out;

  const size_t NE = (size_t)16 * 256 * 4096;  // 16,777,216
  float* wsf  = (float*)d_ws;
  float* XG   = wsf;
  float* G1   = wsf + NE;       // becomes x_low after k3
  float* GAP  = wsf + 2 * NE;   // 4096
  float* GSUM = GAP + 4096;     // 256
  float* GSQ  = GSUM + 256;     // 256
  float* RW   = GSQ + 256;      // 4096
  float* SCA  = RW + 4096;      // 4096
  float* BNS  = SCA + 4096;     // 256
  float* BNB  = BNS + 256;      // 256
  float* LOWS = BNB + 256;      // 4096
  float* GATE = LOWS + 4096;    // 4096
  float* MUG  = GATE + 4096;    // 256
  float* ISD  = MUG + 256;      // 256
  float* XH   = ISD + 256;      // 262144
  float* XW   = XH + 262144;    // 262144
  float* SH   = XW + 262144;    // 262144
  float* SW   = SH + 262144;    // 262144

  kz_zero<<<(4608 + 255) / 256, 256, 0, stream>>>(GAP, 4608);
  k1_gemm12<<<2048, 256, 0, stream>>>(x1, x2, c1w, c1b, c2w, c2b, G1, XG, GAP, GSUM, GSQ);
  k2_small<<<16, 256, 0, stream>>>(GAP, GSUM, GSQ, caw, bng, bnb, bnrm, bnrv,
                                   RW, SCA, BNS, BNB, MUG, ISD);
  k3_masks<<<4096, 256, 0, stream>>>(XG, G1, RW, BNS, BNB, LOWS);
  k4_gate<<<16, 256, 0, stream>>>(LOWS, gatew, gateb, GATE);
  k5_main<<<2048, 256, 0, stream>>>(XG, G1, c3w, c3b, c4w, c4b, c4gw, c4gb,
                                    dww, dwb, GATE, SCA, MUG, ISD, gamma, beta, out);
  k6_pool<<<4096, 256, 0, stream>>>(out, XH, XW);
  k7_coord<<<16, 256, 0, stream>>>(XH, XW, la1w, lbg, lbb, lbrm, lbrv, fhw, fww, SH, SW);
  k8_final<<<4096, 256, 0, stream>>>(out, SH, SW);
}
